// AttentionBlock_Small_49778670961311
// MI455X (gfx1250) — matrix operand order repaired and checked by reading, not compiled, not run on hardware
//
#include <hip/hip_runtime.h>

// ---------------------------------------------------------------------------
// MI455X (gfx1250) attention block, bf16 WMMA pipeline, fp32 accumulate.
// B=16, C=256, H=W=32 (S=1024), NUM_HEADS=4 (HD=64).
// Async tile staging via GLOBAL_LOAD_ASYNC_TO_LDS_B128 (ASYNCcnt).
// ---------------------------------------------------------------------------

typedef unsigned short u16;
typedef unsigned int   u32;
typedef __attribute__((ext_vector_type(16))) __bf16 v16bf;
typedef __attribute__((ext_vector_type(8)))  float  v8f;

#define WMMA_BF16(a, b, c) \
  __builtin_amdgcn_wmma_f32_16x16x32_bf16(false, (a), false, (b), (short)0, (c), false, false)

union FragU { v16bf v; u32 u[8]; };
union Acc8  { v8f  v; float f[8]; };

__device__ __forceinline__ u16 f2bf(float f) {
  u32 u = __builtin_bit_cast(u32, f);
  u32 r = u + 0x7FFFu + ((u >> 16) & 1u);   // round-to-nearest-even
  return (u16)(r >> 16);
}

// Low 32 bits of a flat pointer into LDS == DS byte offset (ISA 10.2 aperture).
__device__ __forceinline__ u32 lds_addr(const void* p) { return (u32)(size_t)p; }

// 16B global -> LDS async DMA chunk (per-lane addresses), ASYNCcnt-tracked.
__device__ __forceinline__ void async_copy16(u32 ldsoff, const void* gptr) {
  asm volatile("global_load_async_to_lds_b128 %0, %1, off"
               :: "v"(ldsoff), "v"(gptr) : "memory");
}
__device__ __forceinline__ void wait_async0() {
  asm volatile("s_wait_asynccnt 0" ::: "memory");
}

// A-matrix fragment, 16x32 bf16 (ISA 7.12.2): lane m = lane&15; lanes 16-31
// hold the +8 K offset. VGPR j<4 -> K = 8*half + 2j ; j>=4 -> K = 16 + 8*half + 2(j-4).
__device__ __forceinline__ v16bf load_frag_a(const u16* base, int row0, int ld,
                                             int k0, int lane) {
  const int m = lane & 15, lh = lane >> 4;
  FragU fr;
  const u16* p = base + (size_t)(row0 + m) * ld + k0;
#pragma unroll
  for (int j = 0; j < 8; ++j) {
    int k = (j < 4) ? (8 * lh + 2 * j) : (16 + 8 * lh + 2 * (j - 4));
    fr.u[j] = *(const u32*)(p + k);
  }
  return fr.v;
}

// B-matrix fragment, 32x16 bf16, staged in LDS as [N][K].
__device__ __forceinline__ v16bf load_frag_b(const u16* base, int n0, int ld,
                                             int k0, int lane) {
  const int n = lane & 15, lh = lane >> 4;
  FragU fr;
  const u16* p = base + (size_t)(n0 + n) * ld + k0;
#pragma unroll
  for (int j = 0; j < 8; ++j) fr.u[j] = *(const u32*)(p + ((j < 4) ? (8 * lh + 2 * j) : (16 + 8 * lh + 2 * (j - 4))));
  return fr.v;
}

// ---------------------------------------------------------------------------
// fp32 -> bf16 cast
// ---------------------------------------------------------------------------
__global__ __launch_bounds__(256) void cast_f32_to_bf16(const float* __restrict__ in,
                                                        u16* __restrict__ out, int n) {
  int i = blockIdx.x * 256 + threadIdx.x;
  if (i < n) out[i] = f2bf(in[i]);
}

// ---------------------------------------------------------------------------
// Batched GEMM: Out[b] = W[256x256] @ X[b][256x1024], bf16 in, fp32 acc.
// Tile 128x128x32, 256 threads = 8 waves (2x4), each wave 4x2 16x16 frags.
// A-tile staged by async DMA; B-tile transposed to [n][k] via VGPRs.
// MODE 0: bf16 [b][h][s][d] (Q,K) | MODE 1: bf16 [b][h][d][s] (V)
// MODE 2: fp32 [b][c][s] + bias (projection)
// ---------------------------------------------------------------------------
template <int MODE>
__global__ __launch_bounds__(256) void gemm256(const u16* __restrict__ W,
                                               const u16* __restrict__ X,
                                               u16* __restrict__ out16,
                                               float* __restrict__ outf,
                                               const float* __restrict__ bias) {
  __shared__ u16 As[128][40];  // [m][k], 80B rows (16B aligned for B128 DMA)
  __shared__ u16 Bs[128][40];  // [n][k]
  const int tid = threadIdx.x;
  const int bidx = blockIdx.z;
  const int m0 = blockIdx.y * 128, n0 = blockIdx.x * 128;
  const u16* Xb = X + (size_t)bidx * 256 * 1024;
  const int wid = tid >> 5, lane = tid & 31;
  const int wm = (wid >> 2) * 64;  // wave covers 64 rows (4 m-frags)
  const int wn = (wid & 3) * 32;   // wave covers 32 cols (2 n-frags)

  Acc8 acc[4][2] = {};

  for (int kt = 0; kt < 8; ++kt) {
    const int k0 = kt * 32;
    __syncthreads();
    // A tile 128x32: 512 x 16B chunks via async DMA straight into LDS
#pragma unroll
    for (int i = 0; i < 2; ++i) {
      int c = tid + 256 * i;
      int row = c >> 2, u = c & 3;
      async_copy16(lds_addr(&As[row][u * 8]),
                   W + (size_t)(m0 + row) * 256 + k0 + u * 8);
    }
    // B tile 32x128, transposed into [n][k] (2-byte shuffle -> manual)
#pragma unroll
    for (int i = 0; i < 8; ++i) {
      int idx = tid + 256 * i;
      int k = idx >> 6, nu = idx & 63;
      u32 v = *(const u32*)(Xb + (size_t)(k0 + k) * 1024 + n0 + 2 * nu);
      Bs[2 * nu][k]     = (u16)(v & 0xFFFFu);
      Bs[2 * nu + 1][k] = (u16)(v >> 16);
    }
    if (kt + 1 < 8)  // global_prefetch_b8 of next B tile
      __builtin_prefetch(Xb + (size_t)(k0 + 32) * 1024 + n0 + 4 * tid, 0, 1);
    wait_async0();
    __syncthreads();

    v16bf a[4], b[2];
#pragma unroll
    for (int im = 0; im < 4; ++im) a[im] = load_frag_a(&As[0][0], wm + 16 * im, 40, 0, lane);
#pragma unroll
    for (int jn = 0; jn < 2; ++jn) b[jn] = load_frag_b(&Bs[0][0], wn + 16 * jn, 40, 0, lane);
#pragma unroll
    for (int im = 0; im < 4; ++im)
#pragma unroll
      for (int jn = 0; jn < 2; ++jn)
        acc[im][jn].v = WMMA_BF16(a[im], b[jn], acc[im][jn].v);
  }

  // C/D layout: VGPR r holds M = r (lanes 0-15) / M = r+8 (lanes 16-31), N = lane&15.
  const int lh = lane >> 4, nl = lane & 15;
#pragma unroll
  for (int im = 0; im < 4; ++im)
#pragma unroll
    for (int jn = 0; jn < 2; ++jn)
#pragma unroll
      for (int r = 0; r < 8; ++r) {
        const int o = m0 + wm + 16 * im + r + 8 * lh;  // out channel
        const int s = n0 + wn + 16 * jn + nl;          // spatial
        const float val = acc[im][jn].f[r];
        if (MODE == 0) {
          out16[(((size_t)bidx * 4 + (o >> 6)) * 1024 + s) * 64 + (o & 63)] = f2bf(val);
        } else if (MODE == 1) {
          out16[(((size_t)bidx * 4 + (o >> 6)) * 64 + (o & 63)) * 1024 + s] = f2bf(val);
        } else {
          outf[((size_t)bidx * 256 + o) * 1024 + s] = val + bias[o];
        }
      }
}

// ---------------------------------------------------------------------------
// Flash attention per (b,head): S=1024, hd=64. 128 threads = 4 waves, each
// wave owns 16 query rows; 64-key tiles with online softmax. scale = 1/16.
// Q,K: [b][h][s][d] ; V: [b][h][d][s] ; out: [b][c][s] bf16.
// All tile staging via async DMA to LDS.
// ---------------------------------------------------------------------------
__global__ __launch_bounds__(128) void attn_flash(const u16* __restrict__ Q,
                                                  const u16* __restrict__ K,
                                                  const u16* __restrict__ V,
                                                  u16* __restrict__ Out) {
  __shared__ u16 Qs[64][72];  // [q_row][d], 144B rows (16B aligned)
  __shared__ u16 Ks[64][72];  // [t][d]   == B-stage [N][K] for Q.K^T
  __shared__ u16 Vs[64][72];  // [d][t]   == B-stage [N][K] for P.V
  __shared__ u16 Ps[64][72];  // [q_row][t], per-wave disjoint rows

  const int tid = threadIdx.x;
  const int bh = blockIdx.y;  // b*4 + h
  const int q0 = blockIdx.x * 64;
  const int w = tid >> 5, lane = tid & 31;
  const int lh = lane >> 4, nl = lane & 15;
  const size_t qk_base = (size_t)bh * 1024 * 64;
  const size_t v_base  = (size_t)bh * 64 * 1024;

  // Q block (64x64 bf16 = 512 x 16B chunks) via async DMA
#pragma unroll
  for (int i = 0; i < 4; ++i) {
    int c = tid + 128 * i;
    int row = c >> 3, u = c & 7;
    async_copy16(lds_addr(&Qs[row][u * 8]),
                 Q + qk_base + (size_t)(q0 + row) * 64 + u * 8);
  }

  Acc8 Oa[4] = {};
  float rm[8], rl[8];
#pragma unroll
  for (int r = 0; r < 8; ++r) { rm[r] = -3.0e38f; rl[r] = 0.f; }

  for (int kt = 0; kt < 16; ++kt) {
    const int t0 = kt * 64;
    __syncthreads();  // previous iteration's readers done before refill
#pragma unroll
    for (int i = 0; i < 4; ++i) {
      int c = tid + 128 * i;
      int row = c >> 3, u = c & 7;
      async_copy16(lds_addr(&Ks[row][u * 8]),
                   K + qk_base + (size_t)(t0 + row) * 64 + u * 8);
      async_copy16(lds_addr(&Vs[row][u * 8]),
                   V + v_base + (size_t)row * 1024 + t0 + u * 8);
    }
    wait_async0();   // covers first-pass Q block too
    __syncthreads();

    // Scores: S = (Q Kt) * scale ; A = Q rows, B = K^T staged as [t][d]
    v16bf qa0 = load_frag_a(&Qs[0][0], 16 * w, 72, 0, lane);
    v16bf qa1 = load_frag_a(&Qs[0][0], 16 * w, 72, 32, lane);
    Acc8 sf[4];
#pragma unroll
    for (int j = 0; j < 4; ++j) {
      v8f z = {};
      v16bf kb0 = load_frag_b(&Ks[0][0], 16 * j, 72, 0, lane);
      v16bf kb1 = load_frag_b(&Ks[0][0], 16 * j, 72, 32, lane);
      z = WMMA_BF16(qa0, kb0, z);
      z = WMMA_BF16(qa1, kb1, z);
      sf[j].v = z;
    }
#pragma unroll
    for (int j = 0; j < 4; ++j)
#pragma unroll
      for (int r = 0; r < 8; ++r) sf[j].f[r] *= 0.0625f;

    // Online softmax: lanes 0-15 / 16-31 hold different row groups -> width-16 shuffles.
#pragma unroll
    for (int r = 0; r < 8; ++r) {
      float t = fmaxf(fmaxf(sf[0].f[r], sf[1].f[r]), fmaxf(sf[2].f[r], sf[3].f[r]));
#pragma unroll
      for (int off = 1; off < 16; off <<= 1) t = fmaxf(t, __shfl_xor(t, off, 16));
      float mnew  = fmaxf(rm[r], t);
      float alpha = __expf(rm[r] - mnew);
      rm[r] = mnew;
      float ps = 0.f;
#pragma unroll
      for (int j = 0; j < 4; ++j) {
        float p = __expf(sf[j].f[r] - mnew);
        sf[j].f[r] = p;
        ps += p;
      }
#pragma unroll
      for (int off = 1; off < 16; off <<= 1) ps += __shfl_xor(ps, off, 16);
      rl[r] = rl[r] * alpha + ps;
#pragma unroll
      for (int j = 0; j < 4; ++j) Oa[j].f[r] *= alpha;
    }

    // C-layout -> A-layout conversion of P through LDS (per-wave rows, same-wave
    // DS ops are in order, no barrier needed).
#pragma unroll
    for (int j = 0; j < 4; ++j)
#pragma unroll
      for (int r = 0; r < 8; ++r)
        Ps[16 * w + r + 8 * lh][16 * j + nl] = f2bf(sf[j].f[r]);

    v16bf pa0 = load_frag_a(&Ps[0][0], 16 * w, 72, 0, lane);
    v16bf pa1 = load_frag_a(&Ps[0][0], 16 * w, 72, 32, lane);
#pragma unroll
    for (int jd = 0; jd < 4; ++jd) {
      v16bf vb0 = load_frag_b(&Vs[0][0], 16 * jd, 72, 0, lane);
      Oa[jd].v = WMMA_BF16(pa0, vb0, Oa[jd].v);
      v16bf vb1 = load_frag_b(&Vs[0][0], 16 * jd, 72, 32, lane);
      Oa[jd].v = WMMA_BF16(pa1, vb1, Oa[jd].v);
    }
  }

  // Normalize and store transposed to [b][c][s] bf16 for the proj GEMM.
  const int b = bh >> 2, h = bh & 3;
#pragma unroll
  for (int r = 0; r < 8; ++r) rl[r] = 1.0f / rl[r];
#pragma unroll
  for (int jd = 0; jd < 4; ++jd)
#pragma unroll
    for (int r = 0; r < 8; ++r) {
      int srow = q0 + 16 * w + r + 8 * lh;
      int ch = h * 64 + 16 * jd + nl;
      Out[((size_t)b * 256 + ch) * 1024 + srow] = f2bf(Oa[jd].f[r] * rl[r]);
    }
}

// ---------------------------------------------------------------------------
// Launch
// ---------------------------------------------------------------------------
extern "C" void kernel_launch(void* const* d_in, const int* in_sizes, int n_in,
                              void* d_out, int out_size, void* d_ws, size_t ws_size,
                              hipStream_t stream) {
  (void)in_sizes; (void)n_in; (void)out_size; (void)ws_size;
  const float* x  = (const float*)d_in[0];
  const float* qw = (const float*)d_in[1];
  const float* kw = (const float*)d_in[2];
  const float* vw = (const float*)d_in[3];
  const float* pw = (const float*)d_in[4];
  const float* pb = (const float*)d_in[5];
  float* out = (float*)d_out;

  char* ws = (char*)d_ws;
  size_t off = 0;
  auto alloc = [&](size_t bytes) -> void* {
    void* p = ws + off;
    off += (bytes + 255) & ~(size_t)255;
    return p;
  };
  const size_t XS = (size_t)16 * 256 * 1024;   // elements
  u16* xb   = (u16*)alloc(XS * 2);
  u16* qwb  = (u16*)alloc(256 * 256 * 2);
  u16* kwb  = (u16*)alloc(256 * 256 * 2);
  u16* vwb  = (u16*)alloc(256 * 256 * 2);
  u16* pwb  = (u16*)alloc(256 * 256 * 2);
  u16* Qg   = (u16*)alloc(XS * 2);             // [b][h][s][d]
  u16* Kg   = (u16*)alloc(XS * 2);             // [b][h][s][d]
  u16* Vg   = (u16*)alloc(XS * 2);             // [b][h][d][s]
  u16* attO = (u16*)alloc(XS * 2);             // [b][c][s]

  cast_f32_to_bf16<<<(int)((XS + 255) / 256), 256, 0, stream>>>(x, xb, (int)XS);
  cast_f32_to_bf16<<<256, 256, 0, stream>>>(qw, qwb, 65536);
  cast_f32_to_bf16<<<256, 256, 0, stream>>>(kw, kwb, 65536);
  cast_f32_to_bf16<<<256, 256, 0, stream>>>(vw, vwb, 65536);
  cast_f32_to_bf16<<<256, 256, 0, stream>>>(pw, pwb, 65536);

  dim3 gg(8, 2, 16);  // N tiles, M tiles, batch
  gemm256<0><<<gg, 256, 0, stream>>>(qwb, xb, Qg, nullptr, nullptr);
  gemm256<0><<<gg, 256, 0, stream>>>(kwb, xb, Kg, nullptr, nullptr);
  gemm256<1><<<gg, 256, 0, stream>>>(vwb, xb, Vg, nullptr, nullptr);

  attn_flash<<<dim3(16, 64), 128, 0, stream>>>(Qg, Kg, Vg, attO);

  gemm256<2><<<gg, 256, 0, stream>>>(pwb, attO, nullptr, out, pb);
}
